// QuantizedLinearMXFP4_635655160007
// MI455X (gfx1250) — 16-bit and 8-bit matrix operand order repaired and checked by reading, compiled, not run on hardware
//
#include <hip/hip_runtime.h>

typedef __attribute__((ext_vector_type(16))) __bf16 v16bf;
typedef __attribute__((ext_vector_type(8)))  float  v8f;
typedef __attribute__((ext_vector_type(8)))  int    v8i;

constexpr int Mdim = 8192;
constexpr int Ndim = 4096;
constexpr int Kdim = 4096;
constexpr int QBLK = 32;

#if __has_builtin(__builtin_amdgcn_cvt_pk_fp8_f32)
#define HW_FP8_PK 1
#endif
#if __has_builtin(__builtin_amdgcn_cvt_f32_fp8)
#define HW_FP8_UNPK 1
#endif

__device__ inline unsigned enc_e4m3_sw(float f) {
  unsigned u = __float_as_uint(f);
  unsigned s = (u >> 24) & 0x80u;
  unsigned a = u & 0x7FFFFFFFu;
  if (a >= 0x43E00000u) return s | 0x7Eu;
  int e = (int)(a >> 23) - 127;
  unsigned man = (a & 0x7FFFFFu) | 0x800000u;
  int E = e + 7;
  if (E >= 1) {
    unsigned keep = man >> 20;
    unsigned rem  = man & 0xFFFFFu;
    if (rem > 0x80000u || (rem == 0x80000u && (keep & 1u))) keep++;
    if (keep == 16u) { keep = 8u; E++; }
    if (E > 15 || (E == 15 && (keep & 7u) == 7u)) return s | 0x7Eu;
    return s | ((unsigned)E << 3) | (keep & 7u);
  }
  int shift = 21 - E;
  if (shift > 31) return s;
  unsigned keep = man >> shift;
  unsigned rembits = man & ((1u << shift) - 1u);
  unsigned half = 1u << (shift - 1);
  if (rembits > half || (rembits == half && (keep & 1u))) keep++;
  if (keep >= 8u) return s | (1u << 3);
  return s | keep;
}

__device__ inline float dec_e4m3_sw(unsigned v) {
  unsigned s = (v & 0x80u) << 24;
  unsigned E = (v >> 3) & 0xFu;
  unsigned M = v & 7u;
  float mag = E ? __uint_as_float(((E + 120u) << 23) | (M << 20))
                : (float)M * 0x1p-9f;
  return __uint_as_float(__float_as_uint(mag) | s);
}

__device__ inline unsigned pack4_fp8(float a, float b, float c, float d) {
#ifdef HW_FP8_PK
  int v = __builtin_amdgcn_cvt_pk_fp8_f32(a, b, 0, false);
  v = __builtin_amdgcn_cvt_pk_fp8_f32(c, d, v, true);
  return (unsigned)v;
#else
  return enc_e4m3_sw(a) | (enc_e4m3_sw(b) << 8) |
         (enc_e4m3_sw(c) << 16) | (enc_e4m3_sw(d) << 24);
#endif
}

template <int SEL>
__device__ inline float unpack1_fp8(unsigned pk) {
#ifdef HW_FP8_UNPK
  return __builtin_amdgcn_cvt_f32_fp8((int)pk, SEL);
#else
  return dec_e4m3_sw((pk >> (8 * SEL)) & 0xFFu);
#endif
}

__device__ inline void mxfp4_block_scales(float amax, float& scale, float& inv) {
  amax = fmaxf(amax, 1e-30f);
  const unsigned eb = (__float_as_uint(amax) >> 23) & 0xFFu;
  scale = __uint_as_float((eb - 2u) << 23);
  inv   = __uint_as_float((256u - eb) << 23);
}

__device__ inline float e2m1_nearest(float mag) {
  if      (mag <= 0.25f) return 0.0f;
  else if (mag <= 0.75f) return 0.5f;
  else if (mag <= 1.25f) return 1.0f;
  else if (mag <= 1.75f) return 1.5f;
  else if (mag <= 2.50f) return 2.0f;
  else if (mag <= 3.50f) return 3.0f;
  else if (mag <= 5.00f) return 4.0f;
  else                   return 6.0f;
}

__global__ __launch_bounds__(256) void quant_w_fp8_kernel(
    const float* __restrict__ w, unsigned char* __restrict__ wq) {
  const int gid = blockIdx.x * blockDim.x + threadIdx.x;
  const float4* src = (const float4*)(w + (size_t)gid * QBLK);

  float vals[QBLK];
#pragma unroll
  for (int i = 0; i < 8; ++i) {
    float4 f = src[i];
    vals[i*4+0] = f.x; vals[i*4+1] = f.y; vals[i*4+2] = f.z; vals[i*4+3] = f.w;
  }
  float amax = 0.0f;
#pragma unroll
  for (int i = 0; i < QBLK; ++i) amax = fmaxf(amax, fabsf(vals[i]));
  float scale, inv;
  mxfp4_block_scales(amax, scale, inv);

  float q[QBLK];
#pragma unroll
  for (int i = 0; i < QBLK; ++i) {
    const float mag = fminf(fabsf(vals[i]) * inv, 6.0f);
    q[i] = copysignf(e2m1_nearest(mag) * scale, vals[i]);
  }
  union { unsigned wds[8]; uint4 u[2]; } o;
#pragma unroll
  for (int i = 0; i < 8; ++i)
    o.wds[i] = pack4_fp8(q[i*4], q[i*4+1], q[i*4+2], q[i*4+3]);
  uint4* dst = (uint4*)(wq + (size_t)gid * QBLK);
  dst[0] = o.u[0];
  dst[1] = o.u[1];
}

__global__ __launch_bounds__(256) void split_x_fp8_kernel(
    const float* __restrict__ x, unsigned char* __restrict__ xhi,
    unsigned char* __restrict__ xlo) {
  const int gid = blockIdx.x * blockDim.x + threadIdx.x;
  const float4* src = (const float4*)(x + (size_t)gid * 64);

  float v[64];
#pragma unroll
  for (int i = 0; i < 16; ++i) {
    float4 f = src[i];
    v[i*4+0] = f.x; v[i*4+1] = f.y; v[i*4+2] = f.z; v[i*4+3] = f.w;
  }

  union { unsigned wds[16]; uint4 u[4]; } hi, lo;
#pragma unroll
  for (int c = 0; c < 8; ++c) {
    const int dw = (c & 1) * 8 + (c >> 1) * 2;
#pragma unroll
    for (int h = 0; h < 2; ++h) {
      const float a = v[c*8 + h*4 + 0], b = v[c*8 + h*4 + 1];
      const float cc = v[c*8 + h*4 + 2], d = v[c*8 + h*4 + 3];
      const unsigned hw = pack4_fp8(a, b, cc, d);
      const float r0 = a  - unpack1_fp8<0>(hw);
      const float r1 = b  - unpack1_fp8<1>(hw);
      const float r2 = cc - unpack1_fp8<2>(hw);
      const float r3 = d  - unpack1_fp8<3>(hw);
      hi.wds[dw + h] = hw;
      lo.wds[dw + h] = pack4_fp8(r0, r1, r2, r3);
    }
  }
  uint4* dh = (uint4*)(xhi + (size_t)gid * 64);
  uint4* dl = (uint4*)(xlo + (size_t)gid * 64);
#pragma unroll
  for (int i = 0; i < 4; ++i) { dh[i] = hi.u[i]; dl[i] = lo.u[i]; }
}

constexpr int BM = 128, BN = 128;
constexpr int BKF = 64;
constexpr int LROW = 80;
constexpr int TILE = 128 * LROW;

__global__ __launch_bounds__(256) void gemm_fp8_wmma_kernel(
    const unsigned char* __restrict__ xhi, const unsigned char* __restrict__ xlo,
    const unsigned char* __restrict__ wq, const float* __restrict__ bias,
    float* __restrict__ out) {
  __shared__ __align__(16) char lds[2 * 3 * TILE];

  const int tid = threadIdx.x;
  const int n0  = blockIdx.x * BN;
  const int m0  = blockIdx.y * BM;
  const int row = tid >> 1;
  const int hlf = tid & 1;

  const int lane = tid & 31;
  const int wv = tid >> 5, wm = wv & 1, wn = wv >> 1;
  const int lr = lane & 15, lh = lane >> 4;

  auto stage = [&](int kt, int buf) {
    const int k0 = kt * BKF;
    char* Ah = lds + buf * 3 * TILE;
    char* Al = Ah + TILE;
    char* Bs = Al + TILE;

    const size_t arow = (size_t)(m0 + row) * Kdim + k0 + hlf * 32;
    const uint4* ph = (const uint4*)(xhi + arow);
    const uint4* pl = (const uint4*)(xlo + arow);
    __builtin_prefetch((const char*)ph + BKF, 0, 1);
    uint4 h0 = ph[0], h1 = ph[1];
    uint4 l0 = pl[0], l1 = pl[1];

    const size_t brow = (size_t)(n0 + row) * Kdim + k0 + hlf * 32;
    const uint4* pb = (const uint4*)(wq + brow);
    uint4 b0 = pb[0], b1 = pb[1];

    uint4* sa = (uint4*)(Ah + row * LROW + hlf * 32);
    sa[0] = h0; sa[1] = h1;
    uint4* sl = (uint4*)(Al + row * LROW + hlf * 32);
    sl[0] = l0; sl[1] = l1;
    uint4* sb = (uint4*)(Bs + row * LROW + hlf * 32);
    sb[0] = b0; sb[1] = b1;
  };

  v8f acc[4][2];
  const v8f zero = {0.f,0.f,0.f,0.f,0.f,0.f,0.f,0.f};
#pragma unroll
  for (int i = 0; i < 4; ++i)
#pragma unroll
    for (int j = 0; j < 2; ++j) acc[i][j] = zero;

  stage(0, 0);

  constexpr int NK = Kdim / BKF;
  for (int kt = 0; kt < NK; ++kt) {
    __syncthreads();
    const int buf = kt & 1;
    if (kt + 1 < NK) stage(kt + 1, buf ^ 1);

    char* Ah = lds + buf * 3 * TILE;
    char* Al = Ah + TILE;
    char* Bs = Al + TILE;

    union Frag { uint4 u[2]; v8i v; };
    Frag b[2];
#pragma unroll
    for (int j = 0; j < 2; ++j) {
      char* p = Bs + (wn * 32 + j * 16 + lr) * LROW;
#pragma unroll
      for (int q = 0; q < 4; ++q)
        ((uint2*)b[j].u)[q] = *(const uint2*)(p + q * 16 + lh * 8);
    }
#pragma unroll
    for (int i = 0; i < 4; ++i) {
      char* pr = Ah + (wm * 64 + i * 16 + lr) * LROW + lh * 32;
      Frag ah, al;
      ah.u[0] = *(const uint4*)(pr);
      ah.u[1] = *(const uint4*)(pr + 16);
      char* pl = Al + (wm * 64 + i * 16 + lr) * LROW + lh * 32;
      al.u[0] = *(const uint4*)(pl);
      al.u[1] = *(const uint4*)(pl + 16);
#pragma unroll
      for (int j = 0; j < 2; ++j) {
        acc[i][j] = __builtin_amdgcn_wmma_f32_16x16x64_fp8_fp8(
            ah.v, b[j].v, (short)0, acc[i][j], false, false);
        acc[i][j] = __builtin_amdgcn_wmma_f32_16x16x64_fp8_fp8(
            al.v, b[j].v, (short)0, acc[i][j], false, false);
      }
    }
  }

#pragma unroll
  for (int j = 0; j < 2; ++j) {
    const int n = n0 + wn * 32 + j * 16 + lr;
    const float bs = bias[n];
#pragma unroll
    for (int i = 0; i < 4; ++i) {
      const int mbase = m0 + wm * 64 + i * 16 + lh * 8;
#pragma unroll
      for (int v = 0; v < 8; ++v)
        out[(size_t)(mbase + v) * Ndim + n] = acc[i][j][v] + bs;
    }
  }
}

__global__ __launch_bounds__(256) void quant_w_bf16_kernel(
    const float* __restrict__ w, __bf16* __restrict__ wq) {
  const int gid = blockIdx.x * blockDim.x + threadIdx.x;
  const float4* src = (const float4*)(w + (size_t)gid * QBLK);
  float vals[QBLK];
#pragma unroll
  for (int i = 0; i < 8; ++i) {
    float4 f = src[i];
    vals[i*4+0] = f.x; vals[i*4+1] = f.y; vals[i*4+2] = f.z; vals[i*4+3] = f.w;
  }
  float amax = 0.0f;
#pragma unroll
  for (int i = 0; i < QBLK; ++i) amax = fmaxf(amax, fabsf(vals[i]));
  float scale, inv;
  mxfp4_block_scales(amax, scale, inv);
  union { __bf16 h[QBLK]; uint4 u[4]; } o;
#pragma unroll
  for (int i = 0; i < QBLK; ++i) {
    const float mag = fminf(fabsf(vals[i]) * inv, 6.0f);
    o.h[i] = (__bf16)copysignf(e2m1_nearest(mag) * scale, vals[i]);
  }
  uint4* dst = (uint4*)(wq + (size_t)gid * QBLK);
#pragma unroll
  for (int i = 0; i < 4; ++i) dst[i] = o.u[i];
}

constexpr int BKB = 32;

__global__ __launch_bounds__(256) void gemm_bf16_wmma_kernel(
    const float* __restrict__ x, const __bf16* __restrict__ wq,
    const float* __restrict__ bias, float* __restrict__ out) {
  __shared__ __align__(16) char lds[4 * TILE];

  const int tid = threadIdx.x;
  const int n0 = blockIdx.x * BN, m0 = blockIdx.y * BM;
  const int row = tid >> 1, hlf = tid & 1;
  const int lane = tid & 31;
  const int wv = tid >> 5, wm = wv & 1, wn = wv >> 1;
  const int lr = lane & 15, lh = lane >> 4;

  auto stage = [&](int kt, int buf) {
    const int k0 = kt * BKB;
    char* As = lds + buf * 2 * TILE;
    char* Bs = As + TILE;
    const float4* px = (const float4*)(x + (size_t)(m0 + row) * Kdim + k0 + hlf * 16);
    float4 f0 = px[0], f1 = px[1], f2 = px[2], f3 = px[3];
    union { __bf16 h[16]; uint4 u[2]; } c;
    c.h[0]=(__bf16)f0.x; c.h[1]=(__bf16)f0.y; c.h[2]=(__bf16)f0.z; c.h[3]=(__bf16)f0.w;
    c.h[4]=(__bf16)f1.x; c.h[5]=(__bf16)f1.y; c.h[6]=(__bf16)f1.z; c.h[7]=(__bf16)f1.w;
    c.h[8]=(__bf16)f2.x; c.h[9]=(__bf16)f2.y; c.h[10]=(__bf16)f2.z; c.h[11]=(__bf16)f2.w;
    c.h[12]=(__bf16)f3.x; c.h[13]=(__bf16)f3.y; c.h[14]=(__bf16)f3.z; c.h[15]=(__bf16)f3.w;
    uint4* pa = (uint4*)(As + row * LROW + hlf * 32);
    pa[0] = c.u[0]; pa[1] = c.u[1];
    const uint4* pw = (const uint4*)(wq + (size_t)(n0 + row) * Kdim + k0 + hlf * 16);
    uint4 b0 = pw[0], b1 = pw[1];
    uint4* pb = (uint4*)(Bs + row * LROW + hlf * 32);
    pb[0] = b0; pb[1] = b1;
  };

  v8f acc[4][2];
  const v8f zero = {0.f,0.f,0.f,0.f,0.f,0.f,0.f,0.f};
#pragma unroll
  for (int i = 0; i < 4; ++i)
#pragma unroll
    for (int j = 0; j < 2; ++j) acc[i][j] = zero;

  stage(0, 0);
  constexpr int NK = Kdim / BKB;
  for (int kt = 0; kt < NK; ++kt) {
    __syncthreads();
    const int buf = kt & 1;
    if (kt + 1 < NK) stage(kt + 1, buf ^ 1);
    char* As = lds + buf * 2 * TILE;
    char* Bs = As + TILE;
    union Frag { uint4 u[2]; v16bf v; };
    Frag a[4], b[2];
#pragma unroll
    for (int i = 0; i < 4; ++i) {
      char* p = As + (wm * 64 + i * 16 + lr) * LROW;
      a[i].u[0] = *(const uint4*)(p + lh * 32);
      a[i].u[1] = *(const uint4*)(p + lh * 32 + 16);
    }
#pragma unroll
    for (int j = 0; j < 2; ++j) {
      char* p = Bs + (wn * 32 + j * 16 + lr) * LROW + lh * 32;
      b[j].u[0] = *(const uint4*)(p);
      b[j].u[1] = *(const uint4*)(p + 16);
    }
#pragma unroll
    for (int i = 0; i < 4; ++i)
#pragma unroll
      for (int j = 0; j < 2; ++j)
        acc[i][j] = __builtin_amdgcn_wmma_f32_16x16x32_bf16(
            false, a[i].v, false, b[j].v, (short)0, acc[i][j], false, false);
  }
#pragma unroll
  for (int j = 0; j < 2; ++j) {
    const int n = n0 + wn * 32 + j * 16 + lr;
    const float bs = bias[n];
#pragma unroll
    for (int i = 0; i < 4; ++i) {
      const int mbase = m0 + wm * 64 + i * 16 + lh * 8;
#pragma unroll
      for (int v = 0; v < 8; ++v)
        out[(size_t)(mbase + v) * Ndim + n] = acc[i][j][v] + bs;
    }
  }
}

extern "C" void kernel_launch(void* const* d_in, const int* in_sizes, int n_in,
                              void* d_out, int out_size, void* d_ws, size_t ws_size,
                              hipStream_t stream) {
  const float* x    = (const float*)d_in[0];
  const float* w    = (const float*)d_in[1];
  const float* bias = (const float*)d_in[2];
  float* out = (float*)d_out;

  const size_t WQ8   = (size_t)Ndim * Kdim;
  const size_t PLANE = (size_t)Mdim * Kdim;
  const size_t NEED_FP8 = WQ8 + 2 * PLANE;

  dim3 grid(Ndim / BN, Mdim / BM);

  if (ws_size >= NEED_FP8) {
    unsigned char* wq  = (unsigned char*)d_ws;
    unsigned char* xhi = wq + WQ8;
    unsigned char* xlo = xhi + PLANE;
    quant_w_fp8_kernel<<<(Ndim * Kdim / QBLK) / 256, 256, 0, stream>>>(w, wq);
    split_x_fp8_kernel<<<(int)((PLANE / 64) / 256), 256, 0, stream>>>(x, xhi, xlo);
    gemm_fp8_wmma_kernel<<<grid, 256, 0, stream>>>(xhi, xlo, wq, bias, out);
  } else {
    __bf16* wq = (__bf16*)d_ws;
    quant_w_bf16_kernel<<<(Ndim * Kdim / QBLK) / 256, 256, 0, stream>>>(w, wq);
    gemm_bf16_wmma_kernel<<<grid, 256, 0, stream>>>(x, wq, bias, out);
  }
}
